// GNN_node_87514253623296
// MI455X (gfx1250) — matrix operand order repaired and checked by reading, not compiled, not run on hardware
//
#include <hip/hip_runtime.h>

#define NN     100000
#define DD     128
#define D2     256
#define EE     600000
#define KK     3
#define LL     3
#define NFEAT  9
#define VOCABN 100
#define BN_EPS 1e-5f

typedef __attribute__((ext_vector_type(16))) __bf16 v16bf;
typedef __attribute__((ext_vector_type(8)))  float  v8f;

union BF16Frag {
  v16bf v;
  __bf16 b[16];
  uint4 u4[2];
};

// ---------------- utility kernels ----------------

__global__ void zero_kernel(float* __restrict__ p, long n) {
  long i = (long)blockIdx.x * blockDim.x + threadIdx.x;
  if (i < n) p[i] = 0.f;
}

__global__ void softmax_kernel(const float* __restrict__ alpha, float* __restrict__ aw) {
  if (blockIdx.x == 0 && threadIdx.x == 0) {
    for (int l = 0; l < LL; ++l) {
      float m = alpha[l * KK];
      for (int k = 1; k < KK; ++k) m = fmaxf(m, alpha[l * KK + k]);
      float e[KK]; float s = 0.f;
      for (int k = 0; k < KK; ++k) { e[k] = __expf(alpha[l * KK + k] - m); s += e[k]; }
      for (int k = 0; k < KK; ++k) aw[l * KK + k] = e[k] / s;
    }
  }
}

// W1 [L][128][256] -> W1t [L][256][128] bf16 ; W2 [L][256][128] -> W2t [L][128][256] bf16
__global__ void convert_w_kernel(const float* __restrict__ W1, const float* __restrict__ W2,
                                 __bf16* __restrict__ W1t, __bf16* __restrict__ W2t) {
  const int per = LL * DD * D2;
  int idx = blockIdx.x * blockDim.x + threadIdx.x;
  if (idx < per) {
    int l = idx / (DD * D2);
    int rem = idx % (DD * D2);
    int n = rem / DD;          // 0..255 output col
    int k = rem % DD;          // 0..127 reduction
    W1t[idx] = (__bf16)W1[(long)l * DD * D2 + (long)k * D2 + n];
  } else if (idx < 2 * per) {
    int j = idx - per;
    int l = j / (DD * D2);
    int rem = j % (DD * D2);
    int n = rem / D2;          // 0..127 output col
    int k = rem % D2;          // 0..255 reduction
    W2t[j] = (__bf16)W2[(long)l * D2 * DD + (long)k * DD + n];
  }
}

// AtomEncoder: h[n,d] = sum_f emb[f, x[n,f], d]
__global__ void embed_kernel(const int* __restrict__ x, const float* __restrict__ emb,
                             float* __restrict__ h) {
  long idx = (long)blockIdx.x * blockDim.x + threadIdx.x;
  if (idx >= (long)NN * DD) return;
  int n = (int)(idx >> 7);
  int d = (int)(idx & (DD - 1));
  float acc = 0.f;
#pragma unroll
  for (int f = 0; f < NFEAT; ++f) {
    int row = x[n * NFEAT + f];
    acc += emb[((long)f * VOCABN + row) * DD + d];
  }
  h[idx] = acc;
}

// segment_sum: agg[dst] += h[src] ; one wave per edge, 4 floats per lane
__global__ void scatter_kernel(const float* __restrict__ h, const int* __restrict__ src,
                               const int* __restrict__ dst, float* __restrict__ agg) {
  long gid = (long)blockIdx.x * blockDim.x + threadIdx.x;
  long e = gid >> 5;
  if (e >= EE) return;
  int d0 = (int)(gid & 31) * 4;
  int s = src[e], t = dst[e];
  float4 v = *reinterpret_cast<const float4*>(h + (long)s * DD + d0);
  float* p = agg + (long)t * DD + d0;
  atomicAdd(p + 0, v.x);
  atomicAdd(p + 1, v.y);
  atomicAdd(p + 2, v.z);
  atomicAdd(p + 3, v.w);
}

// scale/shift from accumulated column sums:  y = (x-mu)*rsqrt(var+eps)*g + b
__global__ void finalize_kernel(const float* __restrict__ gstats, int C,
                                const float* __restrict__ g, const float* __restrict__ b,
                                float* __restrict__ sc, float* __restrict__ sh) {
  int c = blockIdx.x * blockDim.x + threadIdx.x;
  if (c >= C) return;
  float mu  = gstats[c] * (1.0f / (float)NN);
  float var = gstats[C + c] * (1.0f / (float)NN) - mu * mu;
  float rs  = rsqrtf(var + BN_EPS);
  float s   = g[c] * rs;
  sc[c] = s;
  sh[c] = b[c] - mu * s;
}

__global__ void bn_relu_kernel(const float* __restrict__ hin, const float* __restrict__ sc,
                               const float* __restrict__ sh, float* __restrict__ hout) {
  long i = (long)blockIdx.x * blockDim.x + threadIdx.x;
  if (i >= (long)NN * DD) return;
  int c = (int)(i & (DD - 1));
  hout[i] = fmaxf(0.f, hin[i] * sc[c] + sh[c]);
}

// ---------------- GEMM1: z1[N,256] = ((1+eps)h + agg)[N,128] @ W1 ; + column stats ----------------

__global__ __launch_bounds__(256) void gemm1_kernel(
    const float* __restrict__ h, const float* __restrict__ agg,
    const __bf16* __restrict__ Wt,             // [256][128] bf16, K-contig
    const float* __restrict__ eps_p,
    float* __restrict__ z1, float* __restrict__ gstats /* [2*256] */) {
  __shared__ __bf16 sB[128 * DD];              // 32 KB: one N-half (128 cols x 128 K)
  __shared__ float colacc[2 * D2];             // 2 KB

  for (int i = threadIdx.x; i < 2 * D2; i += 256) colacc[i] = 0.f;
  __syncthreads();

  const int wave = threadIdx.x >> 5;
  const int lane = threadIdx.x & 31;
  const int hi   = lane >> 4;
  const int lcol = lane & 15;
  const int row0 = blockIdx.x * 128 + wave * 16;
  const int row  = row0 + (lane & 15);
  const bool valid = row < NN;
  const float epv = 1.0f + eps_p[0];

  // ---- build all 4 K-step A fragments (16x32 bf16 per-wave layout) ----
  BF16Frag af[4];
  {
    const float* hp = h + (long)row * DD;
    const float* ap = agg + (long)row * DD;
#pragma unroll
    for (int kt = 0; kt < 4; ++kt) {
      int kb0 = kt * 32 + hi * 8;
      int kb1 = kb0 + 16;
      float za[16];
      if (valid) {
        float4 a0 = *(const float4*)(hp + kb0);
        float4 a1 = *(const float4*)(hp + kb0 + 4);
        float4 a2 = *(const float4*)(hp + kb1);
        float4 a3 = *(const float4*)(hp + kb1 + 4);
        float4 b0 = *(const float4*)(ap + kb0);
        float4 b1 = *(const float4*)(ap + kb0 + 4);
        float4 b2 = *(const float4*)(ap + kb1);
        float4 b3 = *(const float4*)(ap + kb1 + 4);
        za[0]=epv*a0.x+b0.x;  za[1]=epv*a0.y+b0.y;  za[2]=epv*a0.z+b0.z;  za[3]=epv*a0.w+b0.w;
        za[4]=epv*a1.x+b1.x;  za[5]=epv*a1.y+b1.y;  za[6]=epv*a1.z+b1.z;  za[7]=epv*a1.w+b1.w;
        za[8]=epv*a2.x+b2.x;  za[9]=epv*a2.y+b2.y;  za[10]=epv*a2.z+b2.z; za[11]=epv*a2.w+b2.w;
        za[12]=epv*a3.x+b3.x; za[13]=epv*a3.y+b3.y; za[14]=epv*a3.z+b3.z; za[15]=epv*a3.w+b3.w;
      } else {
#pragma unroll
        for (int j = 0; j < 16; ++j) za[j] = 0.f;
      }
#pragma unroll
      for (int j = 0; j < 16; ++j) af[kt].b[j] = (__bf16)za[j];   // native cvt
    }
  }

  v8f acc[16];
  const v8f vz = {0.f,0.f,0.f,0.f,0.f,0.f,0.f,0.f};
#pragma unroll
  for (int i = 0; i < 16; ++i) acc[i] = vz;

  // ---- two N-halves of B staged through LDS ----
  for (int half = 0; half < 2; ++half) {
    {
      const uint4* g = reinterpret_cast<const uint4*>(Wt + (size_t)half * 128 * DD);
      uint4* s = reinterpret_cast<uint4*>(sB);
      for (int i = threadIdx.x; i < (128 * DD) / 8; i += 256) s[i] = g[i];
    }
    __syncthreads();
#pragma unroll
    for (int nt = 0; nt < 8; ++nt) {
#pragma unroll
      for (int kt = 0; kt < 4; ++kt) {
        BF16Frag bf;
        const uint4* bp = reinterpret_cast<const uint4*>(
            sB + (nt * 16 + lcol) * DD + kt * 32 + hi * 8);
        bf.u4[0] = bp[0];
        bf.u4[1] = bp[2];
        acc[half * 8 + nt] = __builtin_amdgcn_wmma_f32_16x16x32_bf16(
            false, af[kt].v, false, bf.v, (short)0, acc[half * 8 + nt], false, false);
      }
    }
    __syncthreads();
  }

  // ---- store z1 + per-column sum / sumsq (invalid rows contribute exact 0) ----
#pragma unroll
  for (int t = 0; t < 16; ++t) {
    int col = t * 16 + lcol;
    float s = 0.f, q = 0.f;
#pragma unroll
    for (int r = 0; r < 8; ++r) {
      float v = acc[t][r];
      int rr = row0 + hi * 8 + r;
      if (rr < NN) z1[(long)rr * D2 + col] = v;
      s += v;
      q += v * v;
    }
    atomicAdd(&colacc[col], s);
    atomicAdd(&colacc[D2 + col], q);
  }
  __syncthreads();
  for (int i = threadIdx.x; i < 2 * D2; i += 256) atomicAdd(&gstats[i], colacc[i]);
}

// ---------------- GEMM2: hacc = beta*hacc + aw * (relu(bn(z1)) @ W2) ; optional stats ----------------

__global__ __launch_bounds__(256) void gemm2_kernel(
    const float* __restrict__ z1,
    const float* __restrict__ sc1, const float* __restrict__ sh1,
    const __bf16* __restrict__ Wt,             // [128][256] bf16, K-contig
    float* __restrict__ hacc,
    const float* __restrict__ aw_p, int beta, int do_stats,
    float* __restrict__ gstats /* [2*128] */) {
  __shared__ __bf16 sB[64 * D2];               // 32 KB: one N-half (64 cols x 256 K)
  __shared__ float ssc[D2], ssh[D2];
  __shared__ float colacc[2 * DD];

  for (int i = threadIdx.x; i < D2; i += 256) { ssc[i] = sc1[i]; ssh[i] = sh1[i]; }
  for (int i = threadIdx.x; i < 2 * DD; i += 256) colacc[i] = 0.f;
  __syncthreads();

  const int wave = threadIdx.x >> 5;
  const int lane = threadIdx.x & 31;
  const int hi   = lane >> 4;
  const int lcol = lane & 15;
  const int row0 = blockIdx.x * 128 + wave * 16;
  const int row  = row0 + (lane & 15);
  const bool valid = row < NN;
  const float aw = aw_p[0];

  // ---- build all 8 K-step A fragments with fused BN+ReLU ----
  BF16Frag af[8];
  {
    const float* zp = z1 + (long)row * D2;
#pragma unroll
    for (int kt = 0; kt < 8; ++kt) {
      int kb0 = kt * 32 + hi * 8;
      int kb1 = kb0 + 16;
      float za[16];
      if (valid) {
        float4 z0 = *(const float4*)(zp + kb0);
        float4 z1v = *(const float4*)(zp + kb0 + 4);
        float4 z2 = *(const float4*)(zp + kb1);
        float4 z3 = *(const float4*)(zp + kb1 + 4);
        za[0]  = fmaxf(0.f, z0.x  * ssc[kb0+0] + ssh[kb0+0]);
        za[1]  = fmaxf(0.f, z0.y  * ssc[kb0+1] + ssh[kb0+1]);
        za[2]  = fmaxf(0.f, z0.z  * ssc[kb0+2] + ssh[kb0+2]);
        za[3]  = fmaxf(0.f, z0.w  * ssc[kb0+3] + ssh[kb0+3]);
        za[4]  = fmaxf(0.f, z1v.x * ssc[kb0+4] + ssh[kb0+4]);
        za[5]  = fmaxf(0.f, z1v.y * ssc[kb0+5] + ssh[kb0+5]);
        za[6]  = fmaxf(0.f, z1v.z * ssc[kb0+6] + ssh[kb0+6]);
        za[7]  = fmaxf(0.f, z1v.w * ssc[kb0+7] + ssh[kb0+7]);
        za[8]  = fmaxf(0.f, z2.x  * ssc[kb1+0] + ssh[kb1+0]);
        za[9]  = fmaxf(0.f, z2.y  * ssc[kb1+1] + ssh[kb1+1]);
        za[10] = fmaxf(0.f, z2.z  * ssc[kb1+2] + ssh[kb1+2]);
        za[11] = fmaxf(0.f, z2.w  * ssc[kb1+3] + ssh[kb1+3]);
        za[12] = fmaxf(0.f, z3.x  * ssc[kb1+4] + ssh[kb1+4]);
        za[13] = fmaxf(0.f, z3.y  * ssc[kb1+5] + ssh[kb1+5]);
        za[14] = fmaxf(0.f, z3.z  * ssc[kb1+6] + ssh[kb1+6]);
        za[15] = fmaxf(0.f, z3.w  * ssc[kb1+7] + ssh[kb1+7]);
      } else {
#pragma unroll
        for (int j = 0; j < 16; ++j) za[j] = 0.f;
      }
#pragma unroll
      for (int j = 0; j < 16; ++j) af[kt].b[j] = (__bf16)za[j];   // native cvt
    }
  }

  v8f acc[8];
  const v8f vz = {0.f,0.f,0.f,0.f,0.f,0.f,0.f,0.f};
#pragma unroll
  for (int i = 0; i < 8; ++i) acc[i] = vz;

  for (int half = 0; half < 2; ++half) {
    {
      const uint4* g = reinterpret_cast<const uint4*>(Wt + (size_t)half * 64 * D2);
      uint4* s = reinterpret_cast<uint4*>(sB);
      for (int i = threadIdx.x; i < (64 * D2) / 8; i += 256) s[i] = g[i];
    }
    __syncthreads();
#pragma unroll
    for (int nt = 0; nt < 4; ++nt) {
#pragma unroll
      for (int kt = 0; kt < 8; ++kt) {
        BF16Frag bf;
        const uint4* bp = reinterpret_cast<const uint4*>(
            sB + (nt * 16 + lcol) * D2 + kt * 32 + hi * 8);
        bf.u4[0] = bp[0];
        bf.u4[1] = bp[2];
        acc[half * 4 + nt] = __builtin_amdgcn_wmma_f32_16x16x32_bf16(
            false, af[kt].v, false, bf.v, (short)0, acc[half * 4 + nt], false, false);
      }
    }
    __syncthreads();
  }

  // ---- weighted accumulate into hacc; fold stats from the final accumulated value ----
#pragma unroll
  for (int t = 0; t < 8; ++t) {
    int col = t * 16 + lcol;
    float s = 0.f, q = 0.f;
#pragma unroll
    for (int r = 0; r < 8; ++r) {
      int rr = row0 + hi * 8 + r;
      if (rr < NN) {
        float o = aw * acc[t][r];
        if (beta) o += hacc[(long)rr * DD + col];
        hacc[(long)rr * DD + col] = o;
        s += o;
        q += o * o;
      }
    }
    if (do_stats) {
      atomicAdd(&colacc[col], s);
      atomicAdd(&colacc[DD + col], q);
    }
  }
  __syncthreads();
  if (do_stats) {
    for (int i = threadIdx.x; i < 2 * DD; i += 256) atomicAdd(&gstats[i], colacc[i]);
  }
}

// ---------------- host orchestration ----------------

extern "C" void kernel_launch(void* const* d_in, const int* in_sizes, int n_in,
                              void* d_out, int out_size, void* d_ws, size_t ws_size,
                              hipStream_t stream) {
  (void)in_sizes; (void)n_in; (void)out_size; (void)ws_size;
  const int*   x     = (const int*)d_in[0];
  const int*   ke    = (const int*)d_in[1];
  const float* emb   = (const float*)d_in[2];
  const float* W1    = (const float*)d_in[3];
  /* b1 = d_in[4]: exactly cancelled by training-mode BN */
  const float* g1    = (const float*)d_in[5];
  const float* be1   = (const float*)d_in[6];
  const float* W2    = (const float*)d_in[7];
  /* b2 = d_in[8]: exactly cancelled by training-mode BN (softmax weights sum to 1) */
  const float* eps   = (const float*)d_in[9];
  const float* alpha = (const float*)d_in[10];
  const float* bng   = (const float*)d_in[11];
  const float* bnb   = (const float*)d_in[12];
  float* out = (float*)d_out;

  char* ws = (char*)d_ws;
  size_t off = 0;
  auto alloc = [&](size_t bytes) -> void* {
    void* p = ws + off;
    off = (off + bytes + 255) & ~(size_t)255;
    return p;
  };

  float* h    = (float*)alloc((size_t)NN * DD * sizeof(float));
  float* agg  = (float*)alloc((size_t)NN * DD * sizeof(float));
  float* z1   = (float*)alloc((size_t)NN * D2 * sizeof(float));
  float* hacc = (float*)alloc((size_t)NN * DD * sizeof(float));
  __bf16* W1t = (__bf16*)alloc((size_t)LL * DD * D2 * 2);
  __bf16* W2t = (__bf16*)alloc((size_t)LL * DD * D2 * 2);
  float* aw   = (float*)alloc(LL * KK * sizeof(float));
  float* gs   = (float*)alloc(768 * sizeof(float));   // gs1[512] | gs2[256] contiguous
  float* gs1  = gs;
  float* gs2  = gs + 2 * D2;
  float* sc1  = (float*)alloc(D2 * sizeof(float));
  float* sh1  = (float*)alloc(D2 * sizeof(float));
  float* sc2  = (float*)alloc(DD * sizeof(float));
  float* sh2  = (float*)alloc(DD * sizeof(float));

  const int per = LL * DD * D2;
  convert_w_kernel<<<(2 * per + 255) / 256, 256, 0, stream>>>(W1, W2, W1t, W2t);
  softmax_kernel<<<1, 32, 0, stream>>>(alpha, aw);

  const long ND = (long)NN * DD;
  embed_kernel<<<(int)((ND + 255) / 256), 256, 0, stream>>>(x, emb, h);

  const int MB = (NN + 127) / 128;   // 782 row-tiles
  for (int l = 0; l < LL; ++l) {
    for (int k = 0; k < KK; ++k) {
      zero_kernel<<<(int)((ND + 255) / 256), 256, 0, stream>>>(agg, ND);
      zero_kernel<<<3, 256, 0, stream>>>(gs, 768);
      const int* src = ke + ((long)k * 2 + 0) * EE;
      const int* dst = ke + ((long)k * 2 + 1) * EE;
      scatter_kernel<<<(int)(((long)EE * 32 + 255) / 256), 256, 0, stream>>>(h, src, dst, agg);
      gemm1_kernel<<<MB, 256, 0, stream>>>(h, agg, W1t + (size_t)l * DD * D2, eps + l, z1, gs1);
      finalize_kernel<<<1, 256, 0, stream>>>(gs1, D2, g1 + l * D2, be1 + l * D2, sc1, sh1);
      gemm2_kernel<<<MB, 256, 0, stream>>>(z1, sc1, sh1, W2t + (size_t)l * DD * D2, hacc,
                                           aw + l * KK + k, (k > 0) ? 1 : 0,
                                           (k == KK - 1) ? 1 : 0, gs2);
    }
    finalize_kernel<<<1, 128, 0, stream>>>(gs2, DD, bng + l * DD, bnb + l * DD, sc2, sh2);
    bn_relu_kernel<<<(int)((ND + 255) / 256), 256, 0, stream>>>(
        hacc, sc2, sh2, (l == LL - 1) ? out : h);
  }
}
